// CIFModule_45981919871569
// MI455X (gfx1250) — matrix operand order repaired and checked by reading, not compiled, not run on hardware
//
#include <hip/hip_runtime.h>
#include <hip/hip_bf16.h>
#include <math.h>

// Problem dims (fixed by reference)
#define TT 16384
#define HH 1024
#define CC 256
#define KW 3
#define EPSF 1e-5f

typedef __attribute__((ext_vector_type(16))) __bf16 v16bf;
typedef __attribute__((ext_vector_type(8)))  float  v8f;
typedef __attribute__((ext_vector_type(4)))  unsigned int u32x4;
typedef __attribute__((ext_vector_type(4)))  int i32x4;
typedef __attribute__((ext_vector_type(8)))  int i32x8;

#if defined(__gfx1250__) && __has_builtin(__builtin_amdgcn_tensor_load_to_lds)
#define USE_TDM 1
#else
#define USE_TDM 0
#endif

__device__ __forceinline__ unsigned short f32_to_bf16_rne(float f) {
  unsigned int u = __builtin_bit_cast(unsigned int, f);
  u += 0x7FFFu + ((u >> 16) & 1u);
  return (unsigned short)(u >> 16);
}

union FragBF { uint4 u4[2]; v16bf v; };

#if USE_TDM
// Issue one TDM 2D tile load (tile0 x tile1 elements of 2 bytes) into LDS,
// with LDS padding of 4 dwords every 16 dwords (row stride 80B = [.][40]).
__device__ __forceinline__ void tdm_load_2d(unsigned lds_addr,
                                            const unsigned short* gp,
                                            unsigned td0, unsigned td1,
                                            unsigned tile0, unsigned tile1,
                                            unsigned stride0) {
  const unsigned long long ga = (unsigned long long)(uintptr_t)gp;
  u32x4 g0;
  g0[0] = 1u;                                    // count=1, user mode
  g0[1] = lds_addr;                              // lds_addr
  g0[2] = (unsigned)ga;                          // global_addr[31:0]
  g0[3] = (unsigned)((ga >> 32) & 0x1FFFFFFull)  // global_addr[56:32]
          | (2u << 30);                          // type=2 (image)
  i32x8 g1;
  g1[0] = (int)(0x00010000u       // data_size=1 (2B)
                | (1u << 20)      // pad_enable
                | (3u << 22)      // pad_interval: 16 dwords
                | (3u << 25));    // pad_amount: 4 dwords
  g1[1] = (int)(td0 << 16);                       // tensor_dim0 lo16
  g1[2] = (int)((td0 >> 16) | (td1 << 16));       // dim0 hi, tensor_dim1 lo
  g1[3] = (int)((td1 >> 16) | (tile0 << 16));     // dim1 hi, tile_dim0
  g1[4] = (int)(tile1 & 0xFFFFu);                 // tile_dim1 (tile_dim2=0)
  g1[5] = (int)stride0;                           // tensor_dim0_stride lo32
  g1[6] = 0;
  g1[7] = 0;
  const i32x4 gz = {0, 0, 0, 0};
#if __clang_major__ >= 23
  const i32x8 gz8 = {0, 0, 0, 0, 0, 0, 0, 0};
  __builtin_amdgcn_tensor_load_to_lds(g0, g1, gz, gz, gz8, 0);
#else
  __builtin_amdgcn_tensor_load_to_lds(g0, g1, gz, gz, 0);
#endif
}
#endif

// -------------------------------------------------------------------------
// Kernel 0a: rearrange conv_w (C,H,K) fp32 -> wb[k][c][h] bf16
// -------------------------------------------------------------------------
__global__ __launch_bounds__(256)
void convert_w_kernel(const float* __restrict__ w, unsigned short* __restrict__ wb) {
  const int idx = blockIdx.x * 256 + threadIdx.x;   // 0 .. KW*CC*HH-1
  const int k   = idx / (CC * HH);
  const int rem = idx % (CC * HH);
  const int c   = rem / HH;
  const int h   = rem % HH;
  wb[idx] = f32_to_bf16_rne(w[(size_t)c * (HH * KW) + h * KW + k]);
}

// -------------------------------------------------------------------------
// Kernel 0b: encoder_out fp32 -> bf16 with zero guard rows.
//   xbp row r (0..T+1) = x[r-1]; rows 0 and T+1 are zero. This removes all
//   conv boundary handling from the GEMM (tap shift becomes a row offset).
// -------------------------------------------------------------------------
__global__ __launch_bounds__(256)
void convert_x_kernel(const float* __restrict__ enc, unsigned short* __restrict__ xbp) {
  const int idx = blockIdx.x * 256 + threadIdx.x;   // 0 .. (T+2)*H-1
  const int r   = idx >> 10;          // row in padded tensor
  const int h   = idx & (HH - 1);
  unsigned short v = 0;
  if (r >= 1 && r <= TT) v = f32_to_bf16_rne(enc[(size_t)(r - 1) * HH + h]);
  xbp[idx] = v;
}

// -------------------------------------------------------------------------
// Kernel 1: conv-as-GEMM with WMMA bf16 -> f32
//   conv[t,c] = sum_k sum_h x[t+k-1,h] * w[c,h,k] + conv_b[c]
// Block: 256 threads (8 waves), tile M=128 x N=128; wave grid 4(M) x 2(N),
// each wave: 2 M-stripes x 4 N-tiles = 8 WMMAs per K step.
// K loop: 3 taps x (1024/32) = 96 steps of K=32 (bf16 WMMA).
// Double-buffered LDS; both A and B tiles are fetched by the Tensor Data
// Mover (wave 0 issues a descriptor pair for buffer `nxt` right after the
// barrier publishing buffer `cur`, so the TDM engine overlaps the WMMA work).
// -------------------------------------------------------------------------
__global__ __launch_bounds__(256)
void conv_gemm_wmma(const unsigned short* __restrict__ xbp,
                    const unsigned short* __restrict__ wb,
                    const float* __restrict__ cb,
                    float* __restrict__ conv) {
  __shared__ __align__(16) unsigned short As[2][128][40];  // [buf][m][k]
  __shared__ __align__(16) unsigned short Bs[2][128][40];  // [buf][n][k]

  const int t0   = blockIdx.x * 128;
  const int c0   = blockIdx.y * 128;
  const int tid  = threadIdx.x;
  const int lane = tid & 31;
  const int wave = tid >> 5;
  const int wm   = wave & 3;   // 0..3 -> M quarter (32 rows)
  const int wn   = wave >> 2;  // 0..1 -> N half (64 cols)

  v8f acc[2][4] = {};

#if USE_TDM
  const bool issuer = (wave == 0);
  const unsigned as_lds[2] = {(unsigned)(uintptr_t)&As[0][0][0],
                              (unsigned)(uintptr_t)&As[1][0][0]};
  const unsigned bs_lds[2] = {(unsigned)(uintptr_t)&Bs[0][0][0],
                              (unsigned)(uintptr_t)&Bs[1][0][0]};
  if (issuer) {
    // Prologue: fill buffer 0 for ks=0 (tap k=0, h-chunk 0).
    tdm_load_2d(as_lds[0], &xbp[(size_t)t0 * HH], HH, TT + 2, 32, 128, HH);
    tdm_load_2d(bs_lds[0], &wb[(size_t)c0 * HH], HH, CC, 32, 128, HH);
  }
#else
  const int ar = tid >> 1;           // 0..127 row
  const int ac = (tid & 1) * 16;     // 0 or 16
#endif

  for (int ks = 0; ks < 96; ++ks) {
    const int cur = ks & 1;

#if USE_TDM
    if (issuer) __builtin_amdgcn_s_wait_tensorcnt((short)0);
    __syncthreads();   // publish buf cur; all waves done reading buf nxt
    if (ks + 1 < 96 && issuer) {
      const int nxt = cur ^ 1;
      const int kn  = (ks + 1) >> 5;          // next tap
      const int hbn = ((ks + 1) & 31) << 5;   // next h-chunk
      tdm_load_2d(as_lds[nxt], &xbp[(size_t)(t0 + kn) * HH + hbn],
                  HH, TT + 2, 32, 128, HH);
      tdm_load_2d(bs_lds[nxt], &wb[(size_t)kn * CC * HH + (size_t)c0 * HH + hbn],
                  HH, CC, 32, 128, HH);
    }
#else
    const int k  = ks >> 5;
    const int hb = (ks & 31) << 5;
    __syncthreads();
    {  // cooperative bf16 tile loads (fallback path)
      const unsigned short* sa = &xbp[(size_t)(t0 + k + ar) * HH + hb + ac];
      *reinterpret_cast<uint4*>(&As[cur][ar][ac]) =
          *reinterpret_cast<const uint4*>(sa);
      *reinterpret_cast<uint4*>(&As[cur][ar][ac + 8]) =
          *reinterpret_cast<const uint4*>(sa + 8);
      const unsigned short* sb =
          &wb[(size_t)k * CC * HH + (size_t)(c0 + ar) * HH + hb + ac];
      *reinterpret_cast<uint4*>(&Bs[cur][ar][ac]) =
          *reinterpret_cast<const uint4*>(sb);
      *reinterpret_cast<uint4*>(&Bs[cur][ar][ac + 8]) =
          *reinterpret_cast<const uint4*>(sb + 8);
    }
    __syncthreads();
#endif

    // ---- A fragments (16x32 bf16), ISA striping:
    // lane<16: row=lane,    K = {0..7, 16..23}
    // lane>=16: row=lane-16, K = {8..15, 24..31}
    FragBF fa[2];
    const int akb = (lane < 16) ? 0 : 8;
#pragma unroll
    for (int mi = 0; mi < 2; ++mi) {
      const int arow = wm * 32 + mi * 16 + (lane & 15);
      fa[mi].u4[0] = *reinterpret_cast<const uint4*>(&As[cur][arow][akb]);
      fa[mi].u4[1] = *reinterpret_cast<const uint4*>(&As[cur][arow][akb + 16]);
    }

    // ---- B fragments (32x16 bf16): lane<16 -> K {0..7, 16..23}, lane>=16 -> K {8..15, 24..31}
    FragBF fb[4];
    const int bkb = (lane < 16) ? 0 : 8;
#pragma unroll
    for (int sub = 0; sub < 4; ++sub) {
      const int ncol = wn * 64 + sub * 16 + (lane & 15);
      fb[sub].u4[0] = *reinterpret_cast<const uint4*>(&Bs[cur][ncol][bkb]);
      fb[sub].u4[1] = *reinterpret_cast<const uint4*>(&Bs[cur][ncol][bkb + 16]);
    }

    // ---- 8 back-to-back WMMAs ----
#pragma unroll
    for (int mi = 0; mi < 2; ++mi)
#pragma unroll
      for (int sub = 0; sub < 4; ++sub)
        acc[mi][sub] = __builtin_amdgcn_wmma_f32_16x16x32_bf16(
            false, fa[mi].v, false, fb[sub].v, (short)0, acc[mi][sub],
            false, false);
  }

  // ---- epilogue: C/D layout (VGPR v, lane): M = v + (lane>=16 ? 8 : 0),
  // N = lane&15
  const int mrow_off = (lane >> 4) * 8;
#pragma unroll
  for (int mi = 0; mi < 2; ++mi) {
#pragma unroll
    for (int sub = 0; sub < 4; ++sub) {
      const int gc   = c0 + wn * 64 + sub * 16 + (lane & 15);
      const float bv = cb[gc];
#pragma unroll
      for (int v = 0; v < 8; ++v) {
        const int gr = t0 + wm * 32 + mi * 16 + v + mrow_off;
        conv[(size_t)gr * CC + gc] = acc[mi][sub][v] + bv;
      }
    }
  }
}

// -------------------------------------------------------------------------
// Kernel 2: LayerNorm(C=256) + ReLU + Linear(256->1) + sigmoid, one wave/t
// -------------------------------------------------------------------------
__global__ __launch_bounds__(256)
void ln_alpha_kernel(const float* __restrict__ conv,
                     const float* __restrict__ g,
                     const float* __restrict__ b,
                     const float* __restrict__ lw,
                     const float* __restrict__ lb,
                     float* __restrict__ alpha) {
  const int lane = threadIdx.x & 31;
  const int t    = blockIdx.x * 8 + (threadIdx.x >> 5);

  const float4* row = reinterpret_cast<const float4*>(&conv[(size_t)t * CC]);
  const float4 x0 = row[lane * 2];
  const float4 x1 = row[lane * 2 + 1];
  float vals[8] = {x0.x, x0.y, x0.z, x0.w, x1.x, x1.y, x1.z, x1.w};

  float s = 0.f, sq = 0.f;
#pragma unroll
  for (int i = 0; i < 8; ++i) { s += vals[i]; sq += vals[i] * vals[i]; }
#pragma unroll
  for (int off = 16; off >= 1; off >>= 1) {
    s  += __shfl_xor(s, off, 32);
    sq += __shfl_xor(sq, off, 32);
  }
  const float mu   = s * (1.f / CC);
  const float var  = sq * (1.f / CC) - mu * mu;
  const float rstd = rsqrtf(var + EPSF);

  const int c0 = lane * 8;
  float dot = 0.f;
#pragma unroll
  for (int i = 0; i < 8; ++i) {
    const float ln = (vals[i] - mu) * rstd * g[c0 + i] + b[c0 + i];
    dot += fmaxf(ln, 0.f) * lw[c0 + i];
  }
#pragma unroll
  for (int off = 16; off >= 1; off >>= 1) dot += __shfl_xor(dot, off, 32);

  if (lane == 0)
    alpha[t] = 1.f / (1.f + __expf(-(dot + lb[0])));
}

// -------------------------------------------------------------------------
// Kernel 3: CIF scan as prefix sum.
//   acc_t = frac(S_t), fire at t iff floor(S_t) > floor(S_{t-1}).
//   wA[t] = weight of x_t into current segment, wB[t] = spill into next.
// One block, 256 threads x 64-element chunks.
// -------------------------------------------------------------------------
__global__ __launch_bounds__(256)
void cif_scan_kernel(const float* __restrict__ alpha,
                     float* __restrict__ wA, float* __restrict__ wB,
                     int* __restrict__ fire_pos,
                     int* __restrict__ scalars,
                     float* __restrict__ alphas_out) {
  __shared__ float csum[256];
  __shared__ float exs[257];
  const int tid  = threadIdx.x;
  const int base = tid * 64;

  float s = 0.f;
  for (int i = 0; i < 64; ++i) s += alpha[base + i];
  csum[tid] = s;
  __syncthreads();

  if (tid == 0) {
    float r = 0.f;
    for (int i = 0; i < 256; ++i) { exs[i] = r; r += csum[i]; }
    exs[256] = r;
  }
  __syncthreads();

  float S = exs[tid];
  for (int i = 0; i < 64; ++i) {
    const int t   = base + i;
    const float a = alpha[t];
    const float Sp = S;
    S += a;
    const float f0 = floorf(Sp);
    const float f1 = floorf(S);
    if (f1 > f0) {                     // fire: s = acc + a >= 1
      wA[t] = 1.f - (Sp - f0);         // (1 - acc) into current segment
      wB[t] = S - f1;                  // leftover into next segment
      fire_pos[(int)f1 - 1] = t;
    } else {
      wA[t] = a;
      wB[t] = 0.f;
    }
  }

  if (tid == 0) {
    const float total = exs[256];
    const int nf = (int)floorf(total);
    scalars[0] = nf;                           // n_fires
    scalars[1] = (total - (float)nf) > 0.f;    // fin_a > 0 ?
    alphas_out[0] = total;                     // alphas output
  }
}

// -------------------------------------------------------------------------
// Kernel 4: segment gather. Block j builds out[j,:] (H=1024 floats).
//   segment j = [fire_pos[j-1] .. fire_pos[j]] with boundary weights.
// Rows j > n_fires are zeroed (d_out is poisoned by harness).
// -------------------------------------------------------------------------
__global__ __launch_bounds__(256)
void cif_gather_kernel(const float* __restrict__ enc,
                       const float* __restrict__ wA,
                       const float* __restrict__ wB,
                       const int* __restrict__ fire_pos,
                       const int* __restrict__ scalars,
                       float* __restrict__ out) {
  const int j  = blockIdx.x;
  const int h0 = threadIdx.x * 4;
  const int nf = scalars[0];

  float4 acc = {0.f, 0.f, 0.f, 0.f};
  if (j <= nf) {
    const int tb = (j == 0) ? 0 : fire_pos[j - 1];
    const int te = (j < nf) ? fire_pos[j] : (TT - 1);
    int t = tb;
    if (j > 0) {                                  // spill from firing step
      const float w = wB[t];
      const float4 x = *reinterpret_cast<const float4*>(&enc[(size_t)t * HH + h0]);
      acc.x = fmaf(w, x.x, acc.x); acc.y = fmaf(w, x.y, acc.y);
      acc.z = fmaf(w, x.z, acc.z); acc.w = fmaf(w, x.w, acc.w);
      ++t;
    }
    for (; t <= te; ++t) {
      const float w = wA[t];
      const float4 x = *reinterpret_cast<const float4*>(&enc[(size_t)t * HH + h0]);
      acc.x = fmaf(w, x.x, acc.x); acc.y = fmaf(w, x.y, acc.y);
      acc.z = fmaf(w, x.z, acc.z); acc.w = fmaf(w, x.w, acc.w);
    }
    if (j == nf && scalars[1] == 0) {             // fin_a <= 0: drop tail
      acc.x = 0.f; acc.y = 0.f; acc.z = 0.f; acc.w = 0.f;
    }
  }
  *reinterpret_cast<float4*>(&out[(size_t)j * HH + h0]) = acc;
}

// -------------------------------------------------------------------------
// Host launch
// -------------------------------------------------------------------------
extern "C" void kernel_launch(void* const* d_in, const int* in_sizes, int n_in,
                              void* d_out, int out_size, void* d_ws, size_t ws_size,
                              hipStream_t stream) {
  const float* enc    = (const float*)d_in[0];  // (1,T,H)
  const float* conv_w = (const float*)d_in[1];  // (C,H,K)
  const float* conv_b = (const float*)d_in[2];  // (C)
  const float* ln_g   = (const float*)d_in[3];  // (C)
  const float* ln_b   = (const float*)d_in[4];  // (C)
  const float* lin_w  = (const float*)d_in[5];  // (1,C)
  const float* lin_b  = (const float*)d_in[6];  // (1)
  float* out = (float*)d_out;                   // (T+1)*H floats + 1 (alphas)

  // Workspace carve-up (all 256B-aligned by construction)
  char* ws = (char*)d_ws;
  unsigned short* wb  = (unsigned short*)ws; ws += (size_t)KW * CC * HH * 2;       // 1.5 MB
  unsigned short* xbp = (unsigned short*)ws; ws += (size_t)(TT + 2) * HH * 2;      // 32 MB
  float* conv  = (float*)ws;                 ws += (size_t)TT * CC * 4;            // 16 MB
  float* alpha = (float*)ws;                 ws += (size_t)TT * 4;
  float* wA    = (float*)ws;                 ws += (size_t)TT * 4;
  float* wB    = (float*)ws;                 ws += (size_t)TT * 4;
  int*  fire_pos = (int*)ws;                 ws += (size_t)TT * 4;
  int*  scalars  = (int*)ws;                 ws += 256;

  convert_w_kernel<<<(KW * CC * HH) / 256, 256, 0, stream>>>(conv_w, wb);
  convert_x_kernel<<<((TT + 2) * HH) / 256, 256, 0, stream>>>(enc, xbp);
  conv_gemm_wmma<<<dim3(TT / 128, CC / 128), 256, 0, stream>>>(xbp, wb, conv_b, conv);
  ln_alpha_kernel<<<TT / 8, 256, 0, stream>>>(conv, ln_g, ln_b, lin_w, lin_b, alpha);
  cif_scan_kernel<<<1, 256, 0, stream>>>(alpha, wA, wB, fire_pos, scalars,
                                         out + (size_t)(TT + 1) * HH);
  cif_gather_kernel<<<TT + 1, 256, 0, stream>>>(enc, wA, wB, fire_pos, scalars, out);
}
